// PointNet2SemSegMsg_46119358825131
// MI455X (gfx1250) — matrix operand order repaired and checked by reading, not compiled, not run on hardware
//
#include <hip/hip_runtime.h>
#include <hip/hip_bf16.h>

#define Bdim 8
#define Ndim 4096
#define NUM_CLASSES 13

typedef __attribute__((ext_vector_type(16))) _Float16 v16h;
typedef __attribute__((ext_vector_type(8)))  _Float16 v8h;
typedef __attribute__((ext_vector_type(8)))  float    v8f;

// ---------------------------------------------------------------------------
// Pre-pass: X (M x K, f32) -> Xh (Mp x Kp, f16) zero-padded.
// ---------------------------------------------------------------------------
__global__ void convert_pad_x(const float* __restrict__ X, _Float16* __restrict__ Xh,
                              int M, int K, int Kp, long total) {
  long t = (long)blockIdx.x * blockDim.x + threadIdx.x;
  if (t >= total) return;
  const int m = (int)(t / Kp);
  const int k = (int)(t % Kp);
  const float v = (m < M && k < K) ? X[(size_t)m * K + k] : 0.f;
  Xh[t] = (_Float16)v;
}

// W (K x N, f32, row-major) -> Wth (Np x Kp, f16) transposed + zero-padded.
__global__ void transpose_convert_w(const float* __restrict__ W, _Float16* __restrict__ Wt,
                                    int K, int N, int Kp, long total) {
  long t = (long)blockIdx.x * blockDim.x + threadIdx.x;
  if (t >= total) return;
  const int nc = (int)(t / Kp);
  const int k  = (int)(t % Kp);
  const float v = (nc < N && k < K) ? W[(size_t)k * N + nc] : 0.f;
  Wt[t] = (_Float16)v;
}

// ---------------------------------------------------------------------------
// WMMA GEMM: Y[M,N] = Xh[Mp,Kp] * Wth^T + bias.  Guard-free inner loop:
// 4 vector loads + 2 v_wmma_f32_16x16x32_f16 per 32-K chunk.
// Block = 128 threads = 4 waves; wave w computes C tile (16 x 32) at
// (blockIdx.x*4 + w, blockIdx.y).
// ---------------------------------------------------------------------------
__global__ void gemm_bias_wmma(const _Float16* __restrict__ Xh,
                               const _Float16* __restrict__ Wth,
                               const float* __restrict__ bias,
                               float* __restrict__ Y,
                               int M, int N, int Kp, int Mp) {
  const int w      = threadIdx.x >> 5;
  const int tile_m = blockIdx.x * 4 + w;
  if (tile_m * 16 >= Mp) return;
  const int tile_n = blockIdx.y;
  const int lane   = threadIdx.x & 31;
  const int half   = lane >> 4;
  const int l16    = lane & 15;

  const _Float16* Arow  = Xh  + (size_t)(tile_m * 16 + l16) * Kp;
  const _Float16* Brow0 = Wth + (size_t)(tile_n * 32 + l16) * Kp;
  const _Float16* Brow1 = Brow0 + (size_t)16 * Kp;

  v8f acc0 = {0.f, 0.f, 0.f, 0.f, 0.f, 0.f, 0.f, 0.f};
  v8f acc1 = {0.f, 0.f, 0.f, 0.f, 0.f, 0.f, 0.f, 0.f};

  for (int k0 = 0; k0 < Kp; k0 += 32) {
    // A fragment (16x32 f16): 16 contiguous halves per lane (k = 16*half + s).
    const v8h alo = *(const v8h*)(Arow + k0 + 16 * half);
    const v8h ahi = *(const v8h*)(Arow + k0 + 16 * half + 8);
    const v16h a = __builtin_shufflevector(alo, ahi, 0, 1, 2, 3, 4, 5, 6, 7,
                                           8, 9, 10, 11, 12, 13, 14, 15);
    // B fragments (32x16 f16): 16 contiguous halves per lane.
    const v16h b0 = *(const v16h*)(Brow0 + k0 + 16 * half);
    const v16h b1 = *(const v16h*)(Brow1 + k0 + 16 * half);
    acc0 = __builtin_amdgcn_wmma_f32_16x16x32_f16(false, a, false, b0,
                                                  (short)0, acc0, false, false);
    acc1 = __builtin_amdgcn_wmma_f32_16x16x32_f16(false, a, false, b1,
                                                  (short)0, acc1, false, false);
  }

  // Epilogue: C/D layout -> VGPR r holds M = r + 8*half, lane&15 = column.
  const int col0 = tile_n * 32 + l16;
  const int col1 = col0 + 16;
  const float bv0 = (bias && col0 < N) ? bias[col0] : 0.f;
  const float bv1 = (bias && col1 < N) ? bias[col1] : 0.f;
  #pragma unroll
  for (int r = 0; r < 8; ++r) {
    const int m = tile_m * 16 + r + 8 * half;
    if (m < M) {
      if (col0 < N) Y[(size_t)m * N + col0] = acc0[r] + bv0;
      if (col1 < N) Y[(size_t)m * N + col1] = acc1[r] + bv1;
    }
  }
}

// ---------------------------------------------------------------------------
// BatchNorm helpers (population mean/var over all rows, per channel)
// ---------------------------------------------------------------------------
__global__ void zero_kernel(float* p, int n) {
  int t = blockIdx.x * blockDim.x + threadIdx.x;
  if (t < n) p[t] = 0.f;
}

__global__ void bn_reduce_kernel(const float* __restrict__ Y, float* __restrict__ stats,
                                 long M, int C) {
  __shared__ float sh[1024];  // [0,512): sum, [512,1024): sumsq  (C <= 512)
  for (int i = threadIdx.x; i < 1024; i += blockDim.x) sh[i] = 0.f;
  __syncthreads();
  const long total = M * (long)C;
  for (long i = (long)blockIdx.x * blockDim.x + threadIdx.x; i < total;
       i += (long)gridDim.x * blockDim.x) {
    const int   c = (int)(i % C);
    const float v = Y[i];
    atomicAdd(&sh[c], v);
    atomicAdd(&sh[512 + c], v * v);
  }
  __syncthreads();
  for (int c = threadIdx.x; c < C; c += blockDim.x) {
    atomicAdd(&stats[c], sh[c]);
    atomicAdd(&stats[C + c], sh[512 + c]);
  }
}

__global__ void bn_apply_kernel(float* __restrict__ Y, const float* __restrict__ stats,
                                const float* __restrict__ gamma,
                                const float* __restrict__ beta, long M, int C) {
  const long total = M * (long)C;
  const float invM = 1.f / (float)M;
  for (long i = (long)blockIdx.x * blockDim.x + threadIdx.x; i < total;
       i += (long)gridDim.x * blockDim.x) {
    const int   c    = (int)(i % C);
    const float mean = stats[c] * invM;
    const float var  = stats[C + c] * invM - mean * mean;
    float v = gamma[c] * (Y[i] - mean) * rsqrtf(var + 1e-5f) + beta[c];
    Y[i] = v > 0.f ? v : 0.f;
  }
}

// ---------------------------------------------------------------------------
// (B,9,N) -> (B*N, 9) transpose
// ---------------------------------------------------------------------------
__global__ void transpose_kernel(const float* __restrict__ feat, float* __restrict__ x0) {
  int t = blockIdx.x * blockDim.x + threadIdx.x;
  const int total = Bdim * Ndim * 9;
  if (t >= total) return;
  const int b = t / (Ndim * 9);
  const int r = t % (Ndim * 9);
  const int n = r / 9;
  const int c = r % 9;
  x0[((size_t)b * Ndim + n) * 9 + c] = feat[((size_t)b * 9 + c) * Ndim + n];
}

// ---------------------------------------------------------------------------
// Farthest point sampling: one block per batch, 256 threads.
// ---------------------------------------------------------------------------
__global__ void fps_kernel(const float* __restrict__ xyz, int n, int ld, int npoint,
                           int* __restrict__ cents) {
  const int b = blockIdx.x;
  const float* P = xyz + (size_t)b * n * ld;
  const int t   = threadIdx.x;
  const int ppt = (n + 255) >> 8;  // <= 16 for n <= 4096
  float dist[16];
  #pragma unroll
  for (int j = 0; j < 16; ++j) dist[j] = 1e10f;

  __shared__ float sv[256];
  __shared__ int   si[256];
  __shared__ int   sfar;
  int far = 0;

  for (int it = 0; it < npoint; ++it) {
    if (t == 0) cents[b * npoint + it] = far;
    const float* cp = P + (size_t)far * ld;
    const float cx = cp[0], cy = cp[1], cz = cp[2];
    float bv = -1.f;
    int   bi = n - 1;
    for (int j = 0; j < ppt; ++j) {
      const int idx = t * ppt + j;
      if (idx < n) {
        const float* p = P + (size_t)idx * ld;
        const float dx = p[0] - cx, dy = p[1] - cy, dz = p[2] - cz;
        const float d = dx * dx + dy * dy + dz * dz;
        float dd = dist[j];
        if (d < dd) dd = d;
        dist[j] = dd;
        if (dd > bv) { bv = dd; bi = idx; }  // strict > keeps first max
      }
    }
    sv[t] = bv; si[t] = bi;
    __syncthreads();
    for (int s = 128; s > 0; s >>= 1) {
      if (t < s) {
        if (sv[t + s] > sv[t] || (sv[t + s] == sv[t] && si[t + s] < si[t])) {
          sv[t] = sv[t + s]; si[t] = si[t + s];
        }
      }
      __syncthreads();
    }
    if (t == 0) sfar = si[0];
    __syncthreads();
    far = sfar;
    __syncthreads();
  }
}

__global__ void gather_cents_kernel(const float* __restrict__ xyz, int n, int ld,
                                    const int* __restrict__ cents, int np,
                                    float* __restrict__ nx) {
  int t = blockIdx.x * blockDim.x + threadIdx.x;
  const int total = Bdim * np;
  if (t >= total) return;
  const int b = t / np;
  const int j = cents[t];
  const float* p = xyz + ((size_t)b * n + j) * ld;
  nx[(size_t)t * 3 + 0] = p[0];
  nx[(size_t)t * 3 + 1] = p[1];
  nx[(size_t)t * 3 + 2] = p[2];
}

// ---------------------------------------------------------------------------
// Ball query: ascending scan, pad with first, replace n-1 with first.
// ---------------------------------------------------------------------------
__global__ void query_ball_kernel(const float* __restrict__ xyz, int ld, int n,
                                  const float* __restrict__ nxyz, int m, float r2,
                                  int k, int* __restrict__ gi) {
  int t = blockIdx.x * blockDim.x + threadIdx.x;
  const int total = Bdim * m;
  if (t >= total) return;
  const int b = t / m;
  const float* P = xyz + (size_t)b * n * ld;
  const float* q = nxyz + (size_t)t * 3;
  const float qx = q[0], qy = q[1], qz = q[2];
  int loc[32];
  int cnt = 0;
  for (int j = 0; j < n && cnt < k; ++j) {
    const float* pj = P + (size_t)j * ld;
    const float dx = pj[0] - qx, dy = pj[1] - qy, dz = pj[2] - qz;
    const float d = dx * dx + dy * dy + dz * dz;
    if (d <= r2) loc[cnt++] = j;
  }
  const int first = (cnt > 0) ? loc[0] : (n - 1);
  int* out = gi + (size_t)t * k;
  for (int s = 0; s < k; ++s) {
    int v = (s < cnt) ? loc[s] : first;
    if (v == n - 1) v = first;
    out[s] = v;
  }
}

// ---------------------------------------------------------------------------
// Grouping: G[(b,p,s), :] = [feats[b,j,:], xyz[b,j]-new_xyz[b,p]]
// ---------------------------------------------------------------------------
__global__ void group_kernel(const float* __restrict__ xyz, int ld, int n,
                             const float* __restrict__ feats, int Cf,
                             const float* __restrict__ nxyz, int m,
                             const int* __restrict__ gi, int k,
                             float* __restrict__ G) {
  int t = blockIdx.x * blockDim.x + threadIdx.x;
  const long total = (long)Bdim * m * k;
  if (t >= total) return;
  const int bp = t / k;
  const int b  = bp / m;
  const int j  = gi[t];
  const float* fr = feats + ((size_t)b * n + j) * Cf;
  float* g = G + (size_t)t * (Cf + 3);
  for (int c = 0; c < Cf; ++c) g[c] = fr[c];
  const float* pj = xyz + ((size_t)b * n + j) * ld;
  const float* q  = nxyz + (size_t)bp * 3;
  g[Cf + 0] = pj[0] - q[0];
  g[Cf + 1] = pj[1] - q[1];
  g[Cf + 2] = pj[2] - q[2];
}

// ---------------------------------------------------------------------------
// Max-pool over the k neighbors; writes into a column slice of Fout.
// ---------------------------------------------------------------------------
__global__ void maxpool_kernel(const float* __restrict__ Yin, int k, int C,
                               float* __restrict__ Fout, int ldF, int colOff,
                               int rows) {
  int t = blockIdx.x * blockDim.x + threadIdx.x;
  const long total = (long)rows * C;
  if (t >= total) return;
  const int bp = t / C;
  const int c  = t % C;
  float mv = -3.0e38f;
  for (int s = 0; s < k; ++s) {
    const float v = Yin[((size_t)bp * k + s) * C + c];
    if (v > mv) mv = v;
  }
  Fout[(size_t)bp * ldF + colOff + c] = mv;
}

// ---------------------------------------------------------------------------
// 3-NN interpolation + concat with skip features.
// ---------------------------------------------------------------------------
__global__ void knn_interp_kernel(const float* __restrict__ xyz1, int ld1, int n1,
                                  const float* __restrict__ xyz2, int ld2, int n2,
                                  const float* __restrict__ f1, int C1,
                                  const float* __restrict__ f2, int C2,
                                  float* __restrict__ X) {
  int t = blockIdx.x * blockDim.x + threadIdx.x;
  const int total = Bdim * n1;
  if (t >= total) return;
  const int b = t / n1;
  const float* q = xyz1 + (size_t)t * ld1;
  const float qx = q[0], qy = q[1], qz = q[2];
  float d0 = 3e38f, d1 = 3e38f, d2 = 3e38f;
  int   i0 = 0, i1 = 0, i2 = 0;
  for (int j = 0; j < n2; ++j) {
    const float* pj = xyz2 + ((size_t)b * n2 + j) * ld2;
    const float dx = pj[0] - qx, dy = pj[1] - qy, dz = pj[2] - qz;
    const float d = dx * dx + dy * dy + dz * dz;
    if (d < d0)      { d2 = d1; i2 = i1; d1 = d0; i1 = i0; d0 = d; i0 = j; }
    else if (d < d1) { d2 = d1; i2 = i1; d1 = d;  i1 = j; }
    else if (d < d2) { d2 = d;  i2 = j; }
  }
  float w0 = 1.f / (d0 + 1e-8f), w1 = 1.f / (d1 + 1e-8f), w2 = 1.f / (d2 + 1e-8f);
  const float ws = w0 + w1 + w2;
  w0 /= ws; w1 /= ws; w2 /= ws;
  float* xr = X + (size_t)t * (C1 + C2);
  if (C1 > 0) {
    const float* fr = f1 + (size_t)t * C1;
    for (int c = 0; c < C1; ++c) xr[c] = fr[c];
  }
  const float* a  = f2 + ((size_t)b * n2 + i0) * C2;
  const float* bb = f2 + ((size_t)b * n2 + i1) * C2;
  const float* cc = f2 + ((size_t)b * n2 + i2) * C2;
  for (int c = 0; c < C2; ++c)
    xr[C1 + c] = w0 * a[c] + w1 * bb[c] + w2 * cc[c];
}

// ---------------------------------------------------------------------------
// Row-wise log-softmax.
// ---------------------------------------------------------------------------
__global__ void log_softmax_kernel(const float* __restrict__ Yin, float* __restrict__ out,
                                   int M, int C) {
  int t = blockIdx.x * blockDim.x + threadIdx.x;
  if (t >= M) return;
  const float* r = Yin + (size_t)t * C;
  float mx = r[0];
  for (int c = 1; c < C; ++c) mx = fmaxf(mx, r[c]);
  float se = 0.f;
  for (int c = 0; c < C; ++c) se += expf(r[c] - mx);
  const float lse = logf(se) + mx;
  for (int c = 0; c < C; ++c) out[(size_t)t * C + c] = r[c] - lse;
}

// ---------------------------------------------------------------------------
// Host orchestration
// ---------------------------------------------------------------------------
struct Layer { const float *W, *b, *g, *be; int cin, cout; };

extern "C" void kernel_launch(void* const* d_in, const int* in_sizes, int n_in,
                              void* d_out, int out_size, void* d_ws, size_t ws_size,
                              hipStream_t stream) {
  (void)n_in; (void)out_size; (void)ws_size;

  const float* features = (const float*)d_in[0];

  // --- param index mapping: handle dict-insertion vs jax-pytree(sorted) ---
  const bool sortedOrder = (in_sizes[1] != 12 * 16);  // sa1.W0 = 192 elems
  int bsa1, bsa2, bsa3, bsa4, bfp4, bfp3, bfp2, bfp1, bhm, bho;
  if (!sortedOrder) {
    bsa1 = 1;  bsa2 = 25; bsa3 = 49; bsa4 = 73;
    bfp4 = 97; bfp3 = 105; bfp2 = 113; bfp1 = 121; bhm = 133; bho = 137;
  } else {
    bfp1 = 1;  bfp2 = 13; bfp3 = 21; bfp4 = 29; bhm = 37; bho = 41;
    bsa1 = 43; bsa2 = 67; bsa3 = 91; bsa4 = 115;
  }
  auto P = [&](int i) -> const float* { return (const float*)d_in[i]; };
  auto mkLayer = [&](int base, int li, int cin, int cout) -> Layer {
    const int o = base + li * 4;
    return Layer{P(o), P(o + 1), P(o + 2), P(o + 3), cin, cout};
  };

  Layer sa1b0[3] = {mkLayer(bsa1, 0, 12, 16),  mkLayer(bsa1, 1, 16, 16),  mkLayer(bsa1, 2, 16, 32)};
  Layer sa1b1[3] = {mkLayer(bsa1 + 12, 0, 12, 32), mkLayer(bsa1 + 12, 1, 32, 32), mkLayer(bsa1 + 12, 2, 32, 64)};
  Layer sa2b0[3] = {mkLayer(bsa2, 0, 99, 64),  mkLayer(bsa2, 1, 64, 64),  mkLayer(bsa2, 2, 64, 128)};
  Layer sa2b1[3] = {mkLayer(bsa2 + 12, 0, 99, 64), mkLayer(bsa2 + 12, 1, 64, 96), mkLayer(bsa2 + 12, 2, 96, 128)};
  Layer sa3b0[3] = {mkLayer(bsa3, 0, 259, 128), mkLayer(bsa3, 1, 128, 196), mkLayer(bsa3, 2, 196, 256)};
  Layer sa3b1[3] = {mkLayer(bsa3 + 12, 0, 259, 128), mkLayer(bsa3 + 12, 1, 128, 196), mkLayer(bsa3 + 12, 2, 196, 256)};
  Layer sa4b0[3] = {mkLayer(bsa4, 0, 515, 256), mkLayer(bsa4, 1, 256, 256), mkLayer(bsa4, 2, 256, 512)};
  Layer sa4b1[3] = {mkLayer(bsa4 + 12, 0, 515, 256), mkLayer(bsa4 + 12, 1, 256, 384), mkLayer(bsa4 + 12, 2, 384, 512)};
  Layer fp4L[2]  = {mkLayer(bfp4, 0, 1536, 256), mkLayer(bfp4, 1, 256, 256)};
  Layer fp3L[2]  = {mkLayer(bfp3, 0, 512, 256),  mkLayer(bfp3, 1, 256, 256)};
  Layer fp2L[2]  = {mkLayer(bfp2, 0, 352, 256),  mkLayer(bfp2, 1, 256, 128)};
  Layer fp1L[3]  = {mkLayer(bfp1, 0, 128, 128),  mkLayer(bfp1, 1, 128, 128), mkLayer(bfp1, 2, 128, 128)};
  Layer Lhm      = mkLayer(bhm, 0, 128, 128);
  const float* Wout = P(bho);
  const float* bout = P(bho + 1);

  // --- workspace bump allocator ---
  char* wp = (char*)d_ws;
  auto carve = [&](size_t bytes) -> void* {
    void* p = (void*)wp;
    wp += (bytes + 255) & ~(size_t)255;
    return p;
  };
  float*    bufA  = (float*)carve(16777216ull * 4);      // 64 MB ping
  float*    bufB  = (float*)carve(16777216ull * 4);      // 64 MB pong
  _Float16* Xh    = (_Float16*)carve(9437184ull * 2);    // padded f16 activations
  _Float16* Wth   = (_Float16*)carve(524288ull * 2);     // padded f16 transposed weights
  float* x0    = (float*)carve((size_t)Bdim * Ndim * 9 * 4);
  float* l1x   = (float*)carve((size_t)Bdim * 1024 * 3 * 4);
  float* l1f   = (float*)carve((size_t)Bdim * 1024 * 96 * 4);
  float* l1f2  = (float*)carve((size_t)Bdim * 1024 * 128 * 4);
  float* l2x   = (float*)carve((size_t)Bdim * 256 * 3 * 4);
  float* l2f   = (float*)carve((size_t)Bdim * 256 * 256 * 4);
  float* l2f2  = (float*)carve((size_t)Bdim * 256 * 256 * 4);
  float* l3x   = (float*)carve((size_t)Bdim * 64 * 3 * 4);
  float* l3f   = (float*)carve((size_t)Bdim * 64 * 512 * 4);
  float* l3f2  = (float*)carve((size_t)Bdim * 64 * 256 * 4);
  float* l4x   = (float*)carve((size_t)Bdim * 16 * 3 * 4);
  float* l4f   = (float*)carve((size_t)Bdim * 16 * 1024 * 4);
  float* l0f   = (float*)carve((size_t)Bdim * Ndim * 128 * 4);
  float* stats = (float*)carve(2048 * 4);
  int*   cents = (int*)carve((size_t)Bdim * 1024 * 4);
  int*   gi    = (int*)carve((size_t)Bdim * 1024 * 32 * 4);

  // --- GEMM wrapper: pad/convert operands, then guard-free WMMA GEMM ---
  auto gemm = [&](const float* X, const float* W, const float* bias, float* Y,
                  long M, int K, int N) {
    const int Mp = (int)((M + 15) / 16 * 16);
    const int Kp = (K + 31) / 32 * 32;
    const int Np = (N + 31) / 32 * 32;
    const long totX = (long)Mp * Kp;
    convert_pad_x<<<(int)((totX + 255) / 256), 256, 0, stream>>>(X, Xh, (int)M, K, Kp, totX);
    const long totW = (long)Np * Kp;
    transpose_convert_w<<<(int)((totW + 255) / 256), 256, 0, stream>>>(W, Wth, K, N, Kp, totW);
    dim3 g((unsigned)((Mp / 16 + 3) / 4), (unsigned)(Np / 32));
    gemm_bias_wmma<<<g, 128, 0, stream>>>(Xh, Wth, bias, Y, (int)M, N, Kp, Mp);
  };

  auto mlp_layer = [&](const float* X, float* Y, const Layer& L, long M) {
    gemm(X, L.W, L.b, Y, M, L.cin, L.cout);
    zero_kernel<<<(2 * L.cout + 255) / 256, 256, 0, stream>>>(stats, 2 * L.cout);
    const long tot = M * (long)L.cout;
    long nb = (tot + 255) / 256;
    int nb1 = (int)(nb > 1024 ? 1024 : nb);
    bn_reduce_kernel<<<nb1, 256, 0, stream>>>(Y, stats, M, L.cout);
    int nb2 = (int)(nb > 65535 ? 65535 : nb);
    bn_apply_kernel<<<nb2, 256, 0, stream>>>(Y, stats, L.g, L.be, M, L.cout);
  };

  auto run_sa = [&](const float* xyz, int ld, const float* feats, int Cf, int n,
                    int np, float* nx, float* outF, int Ctot,
                    float r0, int k0, Layer* Lb0, float r1, int k1, Layer* Lb1) {
    fps_kernel<<<Bdim, 256, 0, stream>>>(xyz, n, ld, np, cents);
    gather_cents_kernel<<<(Bdim * np + 255) / 256, 256, 0, stream>>>(xyz, n, ld, cents, np, nx);
    float rs[2] = {r0, r1};
    int   ks[2] = {k0, k1};
    Layer* Ls[2] = {Lb0, Lb1};
    int col = 0;
    for (int br = 0; br < 2; ++br) {
      const int   k = ks[br];
      const float r = rs[br];
      query_ball_kernel<<<(Bdim * np + 255) / 256, 256, 0, stream>>>(
          xyz, ld, n, nx, np, r * r, k, gi);
      const long rowsG = (long)Bdim * np * k;
      group_kernel<<<(int)((rowsG + 255) / 256), 256, 0, stream>>>(
          xyz, ld, n, feats, Cf, nx, np, gi, k, bufA);
      const float* X = bufA;
      float* Y = bufB;
      for (int li = 0; li < 3; ++li) {
        Y = (li % 2 == 0) ? bufB : bufA;
        mlp_layer(X, Y, Ls[br][li], rowsG);
        X = Y;
      }
      const int Co = Ls[br][2].cout;  // final layer output lives in bufB
      maxpool_kernel<<<(int)(((long)Bdim * np * Co + 255) / 256), 256, 0, stream>>>(
          bufB, k, Co, outF, Ctot, col, Bdim * np);
      col += Co;
    }
  };

  auto run_fp = [&](const float* xyz1, int ld1, int n1, const float* xyz2, int ld2,
                    int n2, const float* f1, int C1, const float* f2, int C2,
                    Layer* Ls, int nL, float* outF) {
    knn_interp_kernel<<<(Bdim * n1 + 255) / 256, 256, 0, stream>>>(
        xyz1, ld1, n1, xyz2, ld2, n2, f1, C1, f2, C2, bufA);
    const long M = (long)Bdim * n1;
    const float* X = bufA;
    for (int i = 0; i < nL; ++i) {
      float* Y = (i == nL - 1) ? outF : ((i % 2 == 0) ? bufB : bufA);
      mlp_layer(X, Y, Ls[i], M);
      X = Y;
    }
  };

  // --- forward pass ---
  transpose_kernel<<<(Bdim * Ndim * 9 + 255) / 256, 256, 0, stream>>>(features, x0);

  run_sa(x0, 9, x0, 9,    Ndim, 1024, l1x, l1f, 96,   0.05f, 16, sa1b0, 0.1f, 32, sa1b1);
  run_sa(l1x, 3, l1f, 96, 1024, 256,  l2x, l2f, 256,  0.1f,  16, sa2b0, 0.2f, 32, sa2b1);
  run_sa(l2x, 3, l2f, 256, 256, 64,   l3x, l3f, 512,  0.2f,  16, sa3b0, 0.4f, 32, sa3b1);
  run_sa(l3x, 3, l3f, 512, 64,  16,   l4x, l4f, 1024, 0.4f,  16, sa4b0, 0.8f, 32, sa4b1);

  run_fp(l3x, 3, 64,   l4x, 3, 16,  l3f, 512, l4f, 1024, fp4L, 2, l3f2);
  run_fp(l2x, 3, 256,  l3x, 3, 64,  l2f, 256, l3f2, 256, fp3L, 2, l2f2);
  run_fp(l1x, 3, 1024, l2x, 3, 256, l1f, 96,  l2f2, 256, fp2L, 2, l1f2);
  run_fp(x0,  9, Ndim, l1x, 3, 1024, nullptr, 0, l1f2, 128, fp1L, 3, l0f);

  // head: shared MLP (128->128 BN+ReLU) then linear 128->13 + log-softmax
  const long Mh = (long)Bdim * Ndim;
  mlp_layer(l0f, bufB, Lhm, Mh);
  gemm(bufB, Wout, bout, bufA, Mh, 128, NUM_CLASSES);
  log_softmax_kernel<<<(int)((Mh + 255) / 256), 256, 0, stream>>>(
      bufA, (float*)d_out, (int)Mh, NUM_CLASSES);
}
